// CustomLoss_58016418234476
// MI455X (gfx1250) — matrix operand order repaired and checked by reading, not compiled, not run on hardware
//
#include <hip/hip_runtime.h>

#define DEV_INLINE __device__ __forceinline__

typedef __attribute__((ext_vector_type(16))) _Float16 v16h;
typedef __attribute__((ext_vector_type(8)))  _Float16 v8h;
typedef __attribute__((ext_vector_type(8)))  float    v8f;

constexpr int   B_ = 32, P_ = 512, T_ = 512, D_ = 768, C_ = 97;
constexpr float THRESH  = 0.8f;
constexpr float NEG_INF = -3.0e38f;

// ---------------------------------------------------------------------------
// Kernel 1: prediction pairs -> unit-normalized f16 rows (head & tail).
// One wave per row. Each lane caches its 24 strided elements per half in
// registers, butterfly-reduces the squared norm (all lanes get it), then
// scales + converts + stores. Hot GEMM loop then needs zero conversions.
// ---------------------------------------------------------------------------
__global__ __launch_bounds__(256) void pred_prep_kernel(
    const float* __restrict__ pairs,
    _Float16* __restrict__ AH, _Float16* __restrict__ AT) {
  const int row  = blockIdx.x * (blockDim.x >> 5) + (threadIdx.x >> 5);
  const int lane = threadIdx.x & 31;
  const float* rp = pairs + (size_t)row * (2 * D_);
  float vh[24], vt[24];
  float sh = 0.f, st = 0.f;
#pragma unroll
  for (int j = 0; j < 24; ++j) {
    const int i = lane + 32 * j;
    float a = rp[i];       vh[j] = a; sh = fmaf(a, a, sh);
    float c = rp[D_ + i];  vt[j] = c; st = fmaf(c, c, st);
  }
#pragma unroll
  for (int m = 16; m >= 1; m >>= 1) {       // butterfly: every lane gets sum
    sh += __shfl_xor(sh, m, 32);
    st += __shfl_xor(st, m, 32);
  }
  const float ih = 1.f / fmaxf(sqrtf(sh), 1e-8f);
  const float it = 1.f / fmaxf(sqrtf(st), 1e-8f);
  _Float16* ah = AH + (size_t)row * D_;
  _Float16* at = AT + (size_t)row * D_;
#pragma unroll
  for (int j = 0; j < 24; ++j) {
    const int i = lane + 32 * j;
    ah[i] = (_Float16)(vh[j] * ih);
    at[i] = (_Float16)(vt[j] * it);
  }
}

// ---------------------------------------------------------------------------
// Kernel 2: gold triplets -> unit-normalized f16 rows (head & tail) + rel id.
// Triplet rows have stride 1537 floats (misaligned); staging fixes alignment.
// ---------------------------------------------------------------------------
__global__ __launch_bounds__(256) void gold_prep_kernel(
    const float* __restrict__ trip,
    _Float16* __restrict__ GH, _Float16* __restrict__ GT,
    int* __restrict__ rel) {
  const int row  = blockIdx.x * (blockDim.x >> 5) + (threadIdx.x >> 5);
  const int lane = threadIdx.x & 31;
  const float* rp = trip + (size_t)row * (2 * D_ + 1);
  float vh[24], vt[24];
  float sh = 0.f, st = 0.f;
#pragma unroll
  for (int j = 0; j < 24; ++j) {
    const int i = lane + 32 * j;
    float a = rp[i];          vh[j] = a; sh = fmaf(a, a, sh);
    float c = rp[D_ + 1 + i]; vt[j] = c; st = fmaf(c, c, st);
  }
#pragma unroll
  for (int m = 16; m >= 1; m >>= 1) {
    sh += __shfl_xor(sh, m, 32);
    st += __shfl_xor(st, m, 32);
  }
  const float ih = 1.f / fmaxf(sqrtf(sh), 1e-8f);
  const float it = 1.f / fmaxf(sqrtf(st), 1e-8f);
  _Float16* gh = GH + (size_t)row * D_;
  _Float16* gt = GT + (size_t)row * D_;
#pragma unroll
  for (int j = 0; j < 24; ++j) {
    const int i = lane + 32 * j;
    gh[i] = (_Float16)(vh[j] * ih);
    gt[i] = (_Float16)(vt[j] * it);
  }
  if (lane == 0) rel[row] = (int)rp[D_];
}

// ---------------------------------------------------------------------------
// A-fragment loader (f16 source): contiguous k order (B uses the same).
// lanes 0-15  hold K = kb+{0..15}
// lanes 16-31 hold K = kb+{16..31}
// One 32B-aligned 16-half run -> one v16h fragment, no conversions.
// ---------------------------------------------------------------------------
DEV_INLINE v16h load_a_frag(const _Float16* __restrict__ row, int kb, int lg) {
  return *(const v16h*)(row + kb + lg * 16);
}

// ---------------------------------------------------------------------------
// Kernel 3: batched dual-GEMM (head & tail cos-sims, inputs pre-normalized)
// + dual-threshold + argmax over T + log-softmax NLL.
// One block = one (batch, 64-row P tile). 8 waves: 4 along M x 2 along N.
// Each wave: 16(M) x 32(N) outputs per T-tile, 4 WMMAs per K step,
// 24 K-steps x 8 T-tiles = 768 WMMAs per wave. EXEC uniform throughout.
// ---------------------------------------------------------------------------
__global__ __launch_bounds__(256) void match_loss_kernel(
    const _Float16* __restrict__ AH, const _Float16* __restrict__ AT,
    const _Float16* __restrict__ GH, const _Float16* __restrict__ GT,
    const float* __restrict__ preds, const int* __restrict__ rel,
    float* __restrict__ partials) {
  __shared__ float sBest[64][2];
  __shared__ int   sIdx[64][2];
  __shared__ float sNll[64];

  const int b   = blockIdx.x >> 3;
  const int p0  = (blockIdx.x & 7) * 64;
  const int tid = threadIdx.x;
  const int w = tid >> 5, lane = tid & 31;
  const int waveM = w >> 1, waveN = w & 1;
  const int lg = lane >> 4, ln = lane & 15;

  // A rows: this lane supplies row m = p0 + waveM*16 + ln of the A fragment.
  const _Float16* Ah = AH + (size_t)(b * P_ + p0 + waveM * 16 + ln) * D_;
  const _Float16* At = AT + (size_t)(b * P_ + p0 + waveM * 16 + ln) * D_;

  float best[8]; int bidx[8];
#pragma unroll
  for (int r = 0; r < 8; ++r) { best[r] = NEG_INF; bidx[r] = 0; }

  for (int t0 = 0; t0 < T_; t0 += 64) {
    const int tBase = t0 + waveN * 32;
    const int tn0 = tBase + ln;        // N-subtile 0 column for this lane
    const int tn1 = tBase + 16 + ln;   // N-subtile 1 column for this lane
    const _Float16* Bh0 = GH + (size_t)(b * T_ + tn0) * D_;
    const _Float16* Bh1 = GH + (size_t)(b * T_ + tn1) * D_;
    const _Float16* Bt0 = GT + (size_t)(b * T_ + tn0) * D_;
    const _Float16* Bt1 = GT + (size_t)(b * T_ + tn1) * D_;

    v8f aH0 = {}, aH1 = {}, aT0 = {}, aT1 = {};
#pragma unroll 2
    for (int kb = 0; kb < D_; kb += 32) {
      v16h fa_h = load_a_frag(Ah, kb, lg);
      v16h fa_t = load_a_frag(At, kb, lg);
      // B fragment, contiguous k as A: lanes 0-15 hold K = kb+0..15, lanes 16-31 K = kb+16..31
      const int ko = kb + lg * 16;
      v16h fb_h0 = *(const v16h*)(Bh0 + ko);
      v16h fb_h1 = *(const v16h*)(Bh1 + ko);
      v16h fb_t0 = *(const v16h*)(Bt0 + ko);
      v16h fb_t1 = *(const v16h*)(Bt1 + ko);
      aH0 = __builtin_amdgcn_wmma_f32_16x16x32_f16(false, fa_h, false, fb_h0, (short)0, aH0, false, false);
      aH1 = __builtin_amdgcn_wmma_f32_16x16x32_f16(false, fa_h, false, fb_h1, (short)0, aH1, false, false);
      aT0 = __builtin_amdgcn_wmma_f32_16x16x32_f16(false, fa_t, false, fb_t0, (short)0, aT0, false, false);
      aT1 = __builtin_amdgcn_wmma_f32_16x16x32_f16(false, fa_t, false, fb_t1, (short)0, aT1, false, false);
    }

    // Accumulators already ARE cosine similarities (inputs pre-normalized).
#pragma unroll
    for (int r = 0; r < 8; ++r) {
      float h0 = aH0[r], v0 = aT0[r];
      float s0 = (h0 > THRESH && v0 > THRESH) ? 0.5f * (h0 + v0) : NEG_INF;
      if (s0 > best[r] || (s0 == best[r] && tn0 < bidx[r])) { best[r] = s0; bidx[r] = tn0; }
      float h1 = aH1[r], v1 = aT1[r];
      float s1 = (h1 > THRESH && v1 > THRESH) ? 0.5f * (h1 + v1) : NEG_INF;
      if (s1 > best[r] || (s1 == best[r] && tn1 < bidx[r])) { best[r] = s1; bidx[r] = tn1; }
    }
  }

  // Reduce argmax over N within each 16-lane group (row constant per group).
#pragma unroll
  for (int r = 0; r < 8; ++r) {
    float v = best[r]; int ti = bidx[r];
#pragma unroll
    for (int m = 1; m <= 8; m <<= 1) {
      float ov = __shfl_xor(v, m, 32);
      int   ot = __shfl_xor(ti, m, 32);
      if (ov > v || (ov == v && ot < ti)) { v = ov; ti = ot; }
    }
    if (ln == 0) {
      const int localRow = waveM * 16 + 8 * lg + r;  // acc elem r -> row
      sBest[localRow][waveN] = v;
      sIdx[localRow][waveN]  = ti;
    }
  }
  __syncthreads();

  // Per-row: combine the two N-waves, pick target class, log-softmax NLL.
  if (tid < 64) {
    float v0 = sBest[tid][0]; int i0 = sIdx[tid][0];
    float v1 = sBest[tid][1]; int i1 = sIdx[tid][1];
    float v; int ti;
    if (v1 > v0 || (v1 == v0 && i1 < i0)) { v = v1; ti = i1; } else { v = v0; ti = i0; }
    int target = (v > -1.0e37f) ? rel[b * T_ + ti] : 0;
    if (target < 0 || target >= C_) target = 0;

    const float* pr = preds + (size_t)(b * P_ + p0 + tid) * C_;
    float mx = pr[0];
#pragma unroll 4
    for (int c = 1; c < C_; ++c) mx = fmaxf(mx, pr[c]);
    float se = 0.f;
#pragma unroll 4
    for (int c = 0; c < C_; ++c) se += __expf(pr[c] - mx);
    sNll[tid] = (mx + __logf(se)) - pr[target];
  }
  __syncthreads();

  if (tid == 0) {
    float s = 0.f;
#pragma unroll 8
    for (int i = 0; i < 64; ++i) s += sNll[i];
    partials[blockIdx.x] = s;   // deterministic: no float atomics
  }
}

// ---------------------------------------------------------------------------
// Kernel 4: deterministic final reduction of 256 block partials -> mean NLL.
// ---------------------------------------------------------------------------
__global__ __launch_bounds__(256) void final_reduce_kernel(
    const float* __restrict__ partials, float* __restrict__ out) {
  __shared__ float s[256];
  const int t = threadIdx.x;
  s[t] = partials[t];
  __syncthreads();
  for (int m = 128; m >= 1; m >>= 1) {
    if (t < m) s[t] += s[t + m];
    __syncthreads();
  }
  if (t == 0) out[0] = s[0] * (1.0f / (float)(B_ * P_));
}

// ---------------------------------------------------------------------------
extern "C" void kernel_launch(void* const* d_in, const int* in_sizes, int n_in,
                              void* d_out, int out_size, void* d_ws, size_t ws_size,
                              hipStream_t stream) {
  (void)in_sizes; (void)n_in; (void)out_size; (void)ws_size;
  const float* pairs = (const float*)d_in[0];   // [B,P,2D]   f32
  const float* preds = (const float*)d_in[1];   // [B,P,C]    f32
  const float* trip  = (const float*)d_in[2];   // [B,T,2D+1] f32
  float* out = (float*)d_out;

  // Workspace (~101 MB): four unit-normalized f16 matrices + rel + partials.
  char* ws = (char*)d_ws;
  size_t off = 0;
  _Float16* AH  = (_Float16*)(ws + off); off += (size_t)B_ * P_ * D_ * sizeof(_Float16);
  _Float16* AT  = (_Float16*)(ws + off); off += (size_t)B_ * P_ * D_ * sizeof(_Float16);
  _Float16* GH  = (_Float16*)(ws + off); off += (size_t)B_ * T_ * D_ * sizeof(_Float16);
  _Float16* GT  = (_Float16*)(ws + off); off += (size_t)B_ * T_ * D_ * sizeof(_Float16);
  int*      rel = (int*)(ws + off);      off += (size_t)B_ * T_ * sizeof(int);
  float*    part = (float*)(ws + off);   off += 256 * sizeof(float);

  pred_prep_kernel<<<(B_ * P_) / 8, 256, 0, stream>>>(pairs, AH, AT);
  gold_prep_kernel<<<(B_ * T_) / 8, 256, 0, stream>>>(trip, GH, GT, rel);
  match_loss_kernel<<<B_ * (P_ / 64), 256, 0, stream>>>(
      AH, AT, GH, GT, preds, rel, part);
  final_reduce_kernel<<<1, 256, 0, stream>>>(part, out);
}
